// GridificationLayer_8864812499080
// MI455X (gfx1250) — matrix operand order repaired and checked by reading, not compiled, not run on hardware
//
#include <hip/hip_runtime.h>
#include <hip/hip_bf16.h>

typedef __bf16 bf16_t;
typedef __attribute__((ext_vector_type(16))) __bf16 v16bf;
typedef __attribute__((ext_vector_type(8)))  float  v8f;

#define HDIM   128
#define NNODES 65536
#define NGRID  32768
#define NEDGES 600000

// ---------------------------------------------------------------------------
// WMMA 16x16x32 bf16 fragment index helpers (CDNA5 ISA 7.12.2, wave32)
// A-matrix 16x32: lane -> row m = lane&15, half = lane>>4 selects K block.
// element e (0..15): pair p=e>>1; K = (p<4 ? 2p : 16+2(p-4)) + 8*half + (e&1)
// For a fixed lane this is two contiguous 8-element (16B) runs.
// ---------------------------------------------------------------------------
__device__ __forceinline__ int a_kpat(int e, int half) {
  int p = e >> 1;
  int base = (p < 4) ? (2 * p) : (16 + 2 * (p - 4));
  return base + half * 8 + (e & 1);
}

__device__ __forceinline__ float silu(float x) {
  return x / (1.0f + __expf(-x));
}

// B fragments pre-swizzled in workspace: fragment (kt,nt) = 512 bf16,
// lane's 16 elements contiguous -> two b128 loads per lane.
__device__ __forceinline__ v16bf load_b_frag(const bf16_t* __restrict__ w, int kt, int nt) {
  int lane = threadIdx.x & 31;
  return *(const v16bf*)(w + ((kt * 8 + nt) * 512) + lane * 16);
}

__device__ __forceinline__ void zero_acc(v8f (&acc)[8]) {
#pragma unroll
  for (int nt = 0; nt < 8; nt++) acc[nt] = (v8f){};
}

// ---------------------------------------------------------------------------
// Register-blocked GEMM: 16 rows x 128 cols, all 8 N-tiles accumulated in
// registers; each A fragment loaded exactly once, 8 WMMAs issued per A.
// ---------------------------------------------------------------------------
template<int KT>
__device__ __forceinline__ void gemm_acc_f32(v8f (&acc)[8], const bf16_t* __restrict__ wsw,
                                             int ktoff, const float* __restrict__ row,
                                             float scale, int half) {
#pragma unroll
  for (int kt = 0; kt < KT; kt++) {
    v16bf a;
#pragma unroll
    for (int e = 0; e < 16; e++) a[e] = (bf16_t)(row[kt * 32 + a_kpat(e, half)] * scale);
#pragma unroll
    for (int nt = 0; nt < 8; nt++) {
      v16bf b = load_b_frag(wsw, kt + ktoff, nt);
      acc[nt] = __builtin_amdgcn_wmma_f32_16x16x32_bf16(false, a, false, b, (short)0,
                                                        acc[nt], false, false);
    }
  }
}

// A rows already bf16 in GLOBAL memory (contiguous 16B runs -> b128 loads)
template<int KT>
__device__ __forceinline__ void gemm_acc_bf16g(v8f (&acc)[8], const bf16_t* __restrict__ wsw,
                                               int ktoff, const bf16_t* __restrict__ row,
                                               int half) {
#pragma unroll
  for (int kt = 0; kt < KT; kt++) {
    v16bf a;
#pragma unroll
    for (int e = 0; e < 16; e++) a[e] = row[kt * 32 + a_kpat(e, half)];
#pragma unroll
    for (int nt = 0; nt < 8; nt++) {
      v16bf b = load_b_frag(wsw, kt + ktoff, nt);
      acc[nt] = __builtin_amdgcn_wmma_f32_16x16x32_bf16(false, a, false, b, (short)0,
                                                        acc[nt], false, false);
    }
  }
}

// A rows bf16 in LDS
template<int KT>
__device__ __forceinline__ void gemm_acc_lds(v8f (&acc)[8], const bf16_t* __restrict__ wsw,
                                             int ktoff, const bf16_t* rowbase, int half) {
#pragma unroll
  for (int kt = 0; kt < KT; kt++) {
    v16bf a;
#pragma unroll
    for (int e = 0; e < 16; e++) a[e] = rowbase[kt * 32 + a_kpat(e, half)];
#pragma unroll
    for (int nt = 0; nt < 8; nt++) {
      v16bf b = load_b_frag(wsw, kt + ktoff, nt);
      acc[nt] = __builtin_amdgcn_wmma_f32_16x16x32_bf16(false, a, false, b, (short)0,
                                                        acc[nt], false, false);
    }
  }
}

// bias + optional SiLU, write C tiles (bf16) into a [16][HDIM] LDS tile
__device__ __forceinline__ void acc_to_hid(const v8f (&acc)[8], const float* __restrict__ bias,
                                           bf16_t (*hid)[HDIM], int lane, bool act) {
  int half = lane >> 4, m = lane & 15;
#pragma unroll
  for (int nt = 0; nt < 8; nt++) {
    float bv = bias[nt * 16 + m];
#pragma unroll
    for (int v = 0; v < 8; v++) {
      float x = acc[nt][v] + bv;
      hid[v + 8 * half][nt * 16 + m] = (bf16_t)(act ? silu(x) : x);
    }
  }
}

// ---------------------------------------------------------------------------
// Weight pre-swizzle: fp32 [kdim x 128] row-major -> bf16 B-fragments.
// B-matrix 32x16 layout: lane -> col n = lane&15, half = lane>>4;
// element e: K = kt*32 + half*8 + (e<8 ? e : e+8). K >= kdim padded with 0.
// ---------------------------------------------------------------------------
__global__ void swizzle_weights(const float* __restrict__ src, bf16_t* __restrict__ dst,
                                int kdim, int ktc) {
  int t = blockIdx.x * blockDim.x + threadIdx.x;
  int total = ktc * 8 * 512;
  if (t >= total) return;
  int frag = t >> 9;
  int idx  = t & 511;
  int lane = idx >> 4, e = idx & 15;
  int kt = frag >> 3, nt = frag & 7;
  int n = lane & 15, half = lane >> 4;
  int k = kt * 32 + half * 8 + (e < 8 ? e : e + 8);
  float v = (k < kdim) ? src[k * HDIM + nt * 16 + n] : 0.0f;
  dst[frag * 512 + lane * 16 + e] = (bf16_t)v;
}

__global__ void zero_f32(float* __restrict__ p, int n) {
  int i = blockIdx.x * blockDim.x + threadIdx.x;
  if (i < n) p[i] = 0.0f;
}

// ---------------------------------------------------------------------------
// Node MLP: nf = silu(x @ w1 + b1) @ w2 + b2, stored bf16 (coalesced via LDS).
// One wave per 16-row tile; 4 waves/block.
// ---------------------------------------------------------------------------
__global__ __launch_bounds__(128)
void node_mlp(const float* __restrict__ x,
              const bf16_t* __restrict__ w1s, const float* __restrict__ b1,
              const bf16_t* __restrict__ w2s, const float* __restrict__ b2,
              bf16_t* __restrict__ nf) {
  __shared__ bf16_t s_hid[4][16][HDIM];
  int w = threadIdx.x >> 5, lane = threadIdx.x & 31;
  int half = lane >> 4, m = lane & 15;
  int tile = blockIdx.x * 4 + w;
  const float* row = x + (size_t)(tile * 16 + m) * HDIM;

  v8f acc[8];
  zero_acc(acc);
  gemm_acc_f32<4>(acc, w1s, 0, row, 1.0f, half);
  acc_to_hid(acc, b1, s_hid[w], lane, true);

  zero_acc(acc);
  gemm_acc_lds<4>(acc, w2s, 0, &s_hid[w][m][0], half);
  acc_to_hid(acc, b2, s_hid[w], lane, false);   // overwrite: layer-1 acts consumed

  // coalesced bf16 store: this lane writes 64 contiguous cols of row m
  uint4* dst = (uint4*)(nf + (size_t)(tile * 16 + m) * HDIM + half * 64);
  const uint4* sp = (const uint4*)(&s_hid[w][m][half * 64]);
#pragma unroll
  for (int i = 0; i < 8; i++) dst[i] = sp[i];
}

// ---------------------------------------------------------------------------
// Fused edge + message kernel: per 16-edge tile
//   edge_attr[16x6] -> em MLP -> ef[16x128] (via LDS transpose)
//   concat(nf[src], ef)[16x256] -> mm MLP; nf half read DIRECTLY from global
//   (A-row of lane l is lane l's own src row -> two b128 loads per fragment)
//   atomic scatter-add of msg and counts by tgt.
// ---------------------------------------------------------------------------
__global__ __launch_bounds__(128)
void edge_msg(const float* __restrict__ npos, const float* __restrict__ gpos,
              const int* __restrict__ eidx,
              const bf16_t* __restrict__ ew1s, const float* __restrict__ eb1,
              const bf16_t* __restrict__ ew2s, const float* __restrict__ eb2,
              const bf16_t* __restrict__ mw1s, const float* __restrict__ mb1,
              const bf16_t* __restrict__ mw2s, const float* __restrict__ mb2,
              const bf16_t* __restrict__ nf,
              float* __restrict__ agg, float* __restrict__ cnt) {
  __shared__ bf16_t s_ef[4][16][HDIM];    // edge-MLP output (C->A transpose)
  __shared__ bf16_t s_hid[4][16][HDIM];
  int w = threadIdx.x >> 5, lane = threadIdx.x & 31;
  int half = lane >> 4, m = lane & 15;
  int e0 = (blockIdx.x * 4 + w) * 16;
  int my_e = e0 + m;
  int src = eidx[my_e];
  int tgt = eidx[NEDGES + my_e];

  // edge_attr A-fragment (K padded 6 -> 32 with zeros)
  float pv[6];
  pv[0] = npos[src * 3 + 0]; pv[1] = npos[src * 3 + 1]; pv[2] = npos[src * 3 + 2];
  pv[3] = gpos[tgt * 3 + 0]; pv[4] = gpos[tgt * 3 + 1]; pv[5] = gpos[tgt * 3 + 2];
  v16bf ea;
#pragma unroll
  for (int e = 0; e < 16; e++) {
    int k = a_kpat(e, half);
    ea[e] = (k < 6) ? (bf16_t)pv[k < 6 ? k : 0] : (bf16_t)0.0f;
  }

  v8f acc[8];

  // em layer 1 (single K-tile, A in registers)
  zero_acc(acc);
#pragma unroll
  for (int nt = 0; nt < 8; nt++) {
    v16bf b = load_b_frag(ew1s, 0, nt);
    acc[nt] = __builtin_amdgcn_wmma_f32_16x16x32_bf16(false, ea, false, b, (short)0,
                                                      acc[nt], false, false);
  }
  acc_to_hid(acc, eb1, s_hid[w], lane, true);

  // em layer 2 -> ef (no activation)
  zero_acc(acc);
  gemm_acc_lds<4>(acc, ew2s, 0, &s_hid[w][m][0], half);
  acc_to_hid(acc, eb2, s_ef[w], lane, false);

  // mm layer 1: K = 256.  K-tiles 0..3 = nf[src] straight from global,
  // K-tiles 4..7 = ef from LDS.
  zero_acc(acc);
  gemm_acc_bf16g<4>(acc, mw1s, 0, nf + (size_t)src * HDIM, half);
  gemm_acc_lds<4>(acc, mw1s, 4, &s_ef[w][m][0], half);
  acc_to_hid(acc, mb1, s_hid[w], lane, true);   // overwrite: em hidden consumed

  // mm layer 2
  zero_acc(acc);
  gemm_acc_lds<4>(acc, mw2s, 0, &s_hid[w][m][0], half);

  // targets for the rows this lane-half holds in the C layout
  int tgts[8];
#pragma unroll
  for (int v = 0; v < 8; v++) tgts[v] = eidx[NEDGES + e0 + v + 8 * half];

  // scatter-add msg + counts
#pragma unroll
  for (int nt = 0; nt < 8; nt++) {
    float bv = mb2[nt * 16 + m];
#pragma unroll
    for (int v = 0; v < 8; v++)
      atomicAdd(&agg[(size_t)tgts[v] * HDIM + nt * 16 + m], acc[nt][v] + bv);
  }
  if (lane < 16) atomicAdd(&cnt[tgt], 1.0f);
}

// ---------------------------------------------------------------------------
// Final update MLP on mean-aggregated grid features (coalesced f32 output).
// ---------------------------------------------------------------------------
__global__ __launch_bounds__(128)
void grid_update(const float* __restrict__ agg, const float* __restrict__ cnt,
                 const bf16_t* __restrict__ w1s, const float* __restrict__ b1,
                 const bf16_t* __restrict__ w2s, const float* __restrict__ b2,
                 float* __restrict__ out) {
  __shared__ bf16_t s_hid[4][16][HDIM];
  __shared__ float  s_out[4][16][HDIM];
  int w = threadIdx.x >> 5, lane = threadIdx.x & 31;
  int half = lane >> 4, m = lane & 15;
  int tile = blockIdx.x * 4 + w;
  int row = tile * 16 + m;
  float c = cnt[row];
  float inv = 1.0f / (c < 1.0f ? 1.0f : c);
  const float* arow = agg + (size_t)row * HDIM;

  v8f acc[8];
  zero_acc(acc);
  gemm_acc_f32<4>(acc, w1s, 0, arow, inv, half);
  acc_to_hid(acc, b1, s_hid[w], lane, true);

  zero_acc(acc);
  gemm_acc_lds<4>(acc, w2s, 0, &s_hid[w][m][0], half);

#pragma unroll
  for (int nt = 0; nt < 8; nt++) {
    float bv = b2[nt * 16 + m];
#pragma unroll
    for (int v = 0; v < 8; v++)
      s_out[w][v + 8 * half][nt * 16 + m] = acc[nt][v] + bv;
  }

  // coalesced f32 store: this lane writes 64 contiguous cols of row m
  uint4* dst = (uint4*)(out + (size_t)row * HDIM + half * 64);
  const uint4* sp = (const uint4*)(&s_out[w][m][half * 64]);
#pragma unroll
  for (int i = 0; i < 16; i++) dst[i] = sp[i];
}

// ---------------------------------------------------------------------------
extern "C" void kernel_launch(void* const* d_in, const int* in_sizes, int n_in,
                              void* d_out, int out_size, void* d_ws, size_t ws_size,
                              hipStream_t stream) {
  (void)in_sizes; (void)n_in; (void)out_size; (void)ws_size;
  const float* node_features = (const float*)d_in[0];
  const float* node_pos      = (const float*)d_in[1];
  const float* grid_pos      = (const float*)d_in[2];
  const int*   edge_index    = (const int*)d_in[3];
  const float* nm_w1 = (const float*)d_in[4];  const float* nm_b1 = (const float*)d_in[5];
  const float* nm_w2 = (const float*)d_in[6];  const float* nm_b2 = (const float*)d_in[7];
  const float* em_w1 = (const float*)d_in[8];  const float* em_b1 = (const float*)d_in[9];
  const float* em_w2 = (const float*)d_in[10]; const float* em_b2 = (const float*)d_in[11];
  const float* mm_w1 = (const float*)d_in[12]; const float* mm_b1 = (const float*)d_in[13];
  const float* mm_w2 = (const float*)d_in[14]; const float* mm_b2 = (const float*)d_in[15];
  const float* um_w1 = (const float*)d_in[16]; const float* um_b1 = (const float*)d_in[17];
  const float* um_w2 = (const float*)d_in[18]; const float* um_b2 = (const float*)d_in[19];

  char* ws = (char*)d_ws;
  bf16_t* wsw = (bf16_t*)ws;
  // fragment-region offsets (units of 512 bf16 per fragment)
  bf16_t* nmw1s = wsw + (size_t)0   * 512;  // 32 frags
  bf16_t* nmw2s = wsw + (size_t)32  * 512;  // 32
  bf16_t* emw1s = wsw + (size_t)64  * 512;  // 8
  bf16_t* emw2s = wsw + (size_t)72  * 512;  // 32
  bf16_t* mmw1s = wsw + (size_t)104 * 512;  // 64
  bf16_t* mmw2s = wsw + (size_t)168 * 512;  // 32
  bf16_t* umw1s = wsw + (size_t)200 * 512;  // 32
  bf16_t* umw2s = wsw + (size_t)232 * 512;  // 32
  bf16_t* nf  = (bf16_t*)(ws + (1u << 20));
  float*  agg = (float*)(ws + (1u << 20) + (size_t)NNODES * HDIM * 2);
  float*  cnt = (float*)((char*)agg + (size_t)NGRID * HDIM * 4);

  // zero agg + cnt (contiguous)
  int nz = NGRID * HDIM + NGRID;
  zero_f32<<<(nz + 255) / 256, 256, 0, stream>>>(agg, nz);

  // swizzle weights to bf16 B-fragment layout
  auto sw = [&](const float* s, bf16_t* d, int kdim, int ktc) {
    int tot = ktc * 8 * 512;
    swizzle_weights<<<(tot + 255) / 256, 256, 0, stream>>>(s, d, kdim, ktc);
  };
  sw(nm_w1, nmw1s, 128, 4); sw(nm_w2, nmw2s, 128, 4);
  sw(em_w1, emw1s, 6, 1);   sw(em_w2, emw2s, 128, 4);
  sw(mm_w1, mmw1s, 256, 8); sw(mm_w2, mmw2s, 128, 4);
  sw(um_w1, umw1s, 128, 4); sw(um_w2, umw2s, 128, 4);

  node_mlp<<<NNODES / 64, 128, 0, stream>>>(node_features, nmw1s, nm_b1, nmw2s, nm_b2, nf);
  edge_msg<<<NEDGES / 64, 128, 0, stream>>>(node_pos, grid_pos, edge_index,
                                            emw1s, em_b1, emw2s, em_b2,
                                            mmw1s, mm_b1, mmw2s, mm_b2,
                                            nf, agg, cnt);
  grid_update<<<NGRID / 64, 128, 0, stream>>>(agg, cnt, umw1s, um_b1, umw2s, um_b2,
                                              (float*)d_out);
}
